// durationLM_79559974191176
// MI455X (gfx1250) — matrix operand order repaired and checked by reading, not compiled, not run on hardware
//
#include <hip/hip_runtime.h>
#include <hip/hip_bf16.h>
#include <math.h>

// Problem dims (fixed by the reference)
#define T_STEPS 256
#define BATCH   64
#define I_DIM   256
#define H_DIM   256
#define N_BR    8
#define F4      1024        // 4*H
#define IH      512         // I+H

typedef __attribute__((ext_vector_type(16))) __bf16 v16bf;
typedef __attribute__((ext_vector_type(8)))  float  v8f;

__device__ __forceinline__ __bf16 f32_to_bf16(float f) {
    union { float f; unsigned u; } v; v.f = f;
    unsigned u = v.u;
    unsigned r = u + 0x7FFFu + ((u >> 16) & 1u);   // round-to-nearest-even
    unsigned short h = (unsigned short)(r >> 16);
    __bf16 out; __builtin_memcpy(&out, &h, 2);
    return out;
}

__device__ __forceinline__ float sigmoidf_(float x) {
    return 1.0f / (1.0f + __expf(-x));
}

// Position within a 32-wide K block: identity, so lane-half h's 16 A elements
// are k = 16h..16h+15 (contiguous order, matching B).
__device__ __forceinline__ int aswz(int km) {
    return km;
}

// ---------------------------------------------------------------------------
// Prep kernels (one-time, fully parallel)
// ---------------------------------------------------------------------------

// W f32 [N, IH, F4] -> Wt bf16 [N, F4, IH]  (WMMA B-column contiguous)
__global__ void prep_weights(const float* __restrict__ w, __bf16* __restrict__ wt) {
    long idx = (long)blockIdx.x * blockDim.x + threadIdx.x;
    const long total = (long)N_BR * IH * F4;
    if (idx >= total) return;
    int n = (int)(idx / ((long)IH * F4));
    long rem = idx - (long)n * IH * F4;
    int k = (int)(rem / F4);
    int f = (int)(rem % F4);
    wt[((long)n * F4 + f) * IH + k] = f32_to_bf16(w[idx]);
}

// x f32 [T, B, I] -> xs bf16 [T, B, I] swizzled into A-fragment order
__global__ void prep_x(const float* __restrict__ x, __bf16* __restrict__ xs) {
    long idx = (long)blockIdx.x * blockDim.x + threadIdx.x;
    const long total = (long)T_STEPS * BATCH * I_DIM;
    if (idx >= total) return;
    int k  = (int)(idx & (I_DIM - 1));
    long rowbase = idx - k;
    int pos = (k & ~31) | aswz(k & 31);
    xs[rowbase + pos] = f32_to_bf16(x[idx]);
}

// bias = bias_i + bias_h  [N, F4]
__global__ void prep_bias(const float* __restrict__ bi, const float* __restrict__ bh,
                          float* __restrict__ bias) {
    int idx = blockIdx.x * blockDim.x + threadIdx.x;
    if (idx < N_BR * F4) bias[idx] = bi[idx] + bh[idx];
}

// zero bf16 h-state and f32 c-state
__global__ void prep_zero(__bf16* __restrict__ hs, float* __restrict__ cst, int count) {
    int idx = blockIdx.x * blockDim.x + threadIdx.x;
    if (idx < count) {
        unsigned short z = 0;
        __builtin_memcpy(&hs[idx], &z, 2);
        cst[idx] = 0.0f;
    }
}

// ---------------------------------------------------------------------------
// One recurrence step. Wave w: branch n = w>>6, M-tile = (w>>4)&3, h-slice = w&15.
// 4 gate tiles accumulated -> gates + cell update + freeze mask in-register.
// K loop fully unrolled with a depth-2 rotating prefetch pipeline (3 buffer
// sets); sched_barrier pins prefetch loads above the WMMAs of each stage.
// ---------------------------------------------------------------------------
__global__ void __launch_bounds__(256)
lstm_step(const __bf16* __restrict__ xs,        // [T, B, I] bf16, A-swizzled
          const int*    __restrict__ dur,       // [T, B]
          const __bf16* __restrict__ wt,        // [N, F4, IH] bf16
          const float*  __restrict__ bias,      // [N, F4]
          __bf16* __restrict__ hs,              // [N, B, H] bf16, A-swizzled
          float*  __restrict__ cst,             // [N, B, H] f32
          float*  __restrict__ out,             // [T, N, B, H] f32
          int t) {
    const int tid   = threadIdx.x;
    const int lane  = tid & 31;
    const int w     = blockIdx.x * 8 + (tid >> 5);     // 0..511
    const int n     = w >> 6;                          // 0..7
    const int mtile = (w >> 4) & 3;                    // 0..3
    const int fg    = w & 15;                          // 0..15

    const int lmod  = lane & 15;
    const int half  = lane >> 4;                       // 0/1 -> K sub-half

    // A-fragment: one contiguous 32B load per K-step (swizzled layout)
    const int arow = mtile * 16 + lmod;
    const __bf16* ax = xs + ((long)t * BATCH + arow) * I_DIM + (half << 4);
    const __bf16* ah = hs + ((long)n * BATCH + arow) * H_DIM + (half << 4);

    // B-fragment rows: column N = lane%16, one per gate
    const int fcol = fg * 16 + lmod;
    const __bf16* wr0 = wt + ((long)n * F4 + 0 * H_DIM + fcol) * IH + (half << 4);
    const __bf16* wr1 = wt + ((long)n * F4 + 1 * H_DIM + fcol) * IH + (half << 4);
    const __bf16* wr2 = wt + ((long)n * F4 + 2 * H_DIM + fcol) * IH + (half << 4);
    const __bf16* wr3 = wt + ((long)n * F4 + 3 * H_DIM + fcol) * IH + (half << 4);

    // Fold bias into accumulator init (bias is column-only -> same for all rows)
    const float bi_ = bias[n * F4 + 0 * H_DIM + fcol];
    const float bf_ = bias[n * F4 + 1 * H_DIM + fcol];
    const float bo_ = bias[n * F4 + 2 * H_DIM + fcol];
    const float bg_ = bias[n * F4 + 3 * H_DIM + fcol];
    v8f acc_i, acc_f, acc_o, acc_g;
    #pragma unroll
    for (int r = 0; r < 8; ++r) { acc_i[r] = bi_; acc_f[r] = bf_; acc_o[r] = bo_; acc_g[r] = bg_; }

    // Static per-kk A source (x for kk<8, recurrent h for kk>=8)
    auto aAddrK = [&](int kk) -> const __bf16* {
        return (kk < 8) ? (ax + kk * 32) : (ah + (kk - 8) * 32);
    };

    // ---- Depth-2 prefetch pipeline over 16 K-steps, 3 rotating buffer sets ----
    v16bf abuf[3];
    v16bf bbuf[3][4];

    #pragma unroll
    for (int p = 0; p < 2; ++p) {                      // prologue: steps 0 and 1
        const int kb = p * 32;
        abuf[p]    = *(const v16bf*)aAddrK(p);
        bbuf[p][0] = *(const v16bf*)(wr0 + kb);
        bbuf[p][1] = *(const v16bf*)(wr1 + kb);
        bbuf[p][2] = *(const v16bf*)(wr2 + kb);
        bbuf[p][3] = *(const v16bf*)(wr3 + kb);
    }

    #pragma unroll
    for (int kk = 0; kk < 16; ++kk) {
        if (kk + 2 < 16) {                             // prefetch step kk+2
            const int s  = (kk + 2) % 3;
            const int kb = (kk + 2) * 32;
            abuf[s]    = *(const v16bf*)aAddrK(kk + 2);
            bbuf[s][0] = *(const v16bf*)(wr0 + kb);
            bbuf[s][1] = *(const v16bf*)(wr1 + kb);
            bbuf[s][2] = *(const v16bf*)(wr2 + kb);
            bbuf[s][3] = *(const v16bf*)(wr3 + kb);
        }
        __builtin_amdgcn_sched_barrier(0);             // keep prefetch above WMMAs

        const int c = kk % 3;
        acc_i = __builtin_amdgcn_wmma_f32_16x16x32_bf16(false, abuf[c], false, bbuf[c][0], (short)0, acc_i, false, false);
        acc_f = __builtin_amdgcn_wmma_f32_16x16x32_bf16(false, abuf[c], false, bbuf[c][1], (short)0, acc_f, true,  false);
        acc_o = __builtin_amdgcn_wmma_f32_16x16x32_bf16(false, abuf[c], false, bbuf[c][2], (short)0, acc_o, true,  false);
        acc_g = __builtin_amdgcn_wmma_f32_16x16x32_bf16(false, abuf[c], false, bbuf[c][3], (short)0, acc_g, true,  false);
        __builtin_amdgcn_sched_barrier(0);             // keep stages in order
    }

    // Epilogue. D layout: VGPR r -> row M = r + 8*half, column N = lane%16.
    const int hcol = fg * 16 + lmod;
    const int hswz = (hcol & ~31) | aswz(hcol & 31);   // lane-constant swizzled h pos

    #pragma unroll
    for (int r = 0; r < 8; ++r) {
        const int brow = mtile * 16 + r + (half << 3);
        const long sbase = ((long)n * BATCH + brow) * H_DIM;

        float ig  = sigmoidf_(acc_i[r]);
        float fgt = sigmoidf_(acc_f[r]);
        float og  = sigmoidf_(acc_o[r]);
        float gg  = tanhf(acc_g[r]);

        float c_old = cst[sbase + hcol];
        int d = dur[t * BATCH + brow];
        bool frozen = n > (d >> 3);                    // RESCALE_RATE = 8
        float c_new = frozen ? c_old : (fgt * c_old + ig * gg);
        float h_new = og * tanhf(c_new);

        cst[sbase + hcol] = c_new;
        hs[sbase + hswz] = f32_to_bf16(h_new);         // bf16, A-swizzled for next step
        out[(((long)t * N_BR + n) * BATCH + brow) * H_DIM + hcol] = h_new;
    }
}

// ---------------------------------------------------------------------------
// Host launcher
// ---------------------------------------------------------------------------
extern "C" void kernel_launch(void* const* d_in, const int* in_sizes, int n_in,
                              void* d_out, int out_size, void* d_ws, size_t ws_size,
                              hipStream_t stream) {
    (void)in_sizes; (void)n_in; (void)out_size; (void)ws_size;

    const float* x    = (const float*)d_in[0];
    const int*   dur  = (const int*)  d_in[1];
    const float* wgt  = (const float*)d_in[2];
    const float* bi   = (const float*)d_in[3];
    const float* bh   = (const float*)d_in[4];
    float*       out  = (float*)d_out;

    // Workspace layout (bytes)
    const size_t WT_BYTES   = (size_t)N_BR * F4 * IH * sizeof(__bf16);          // 8 MB
    const size_t XS_BYTES   = (size_t)T_STEPS * BATCH * I_DIM * sizeof(__bf16); // 8 MB
    const size_t BIAS_BYTES = (size_t)N_BR * F4 * sizeof(float);                // 32 KB
    const size_t HS_BYTES   = (size_t)N_BR * BATCH * H_DIM * sizeof(__bf16);    // 256 KB
    char* ws = (char*)d_ws;
    __bf16* wt   = (__bf16*)(ws);
    __bf16* xs   = (__bf16*)(ws + WT_BYTES);
    float*  bias = (float*) (ws + WT_BYTES + XS_BYTES);
    __bf16* hs   = (__bf16*)(ws + WT_BYTES + XS_BYTES + BIAS_BYTES);
    float*  cst  = (float*) (ws + WT_BYTES + XS_BYTES + BIAS_BYTES + HS_BYTES);

    // One-time prep: weight transpose+bf16, x convert+swizzle, bias fold, zero state
    {
        const long totalW = (long)N_BR * IH * F4;
        prep_weights<<<(int)((totalW + 255) / 256), 256, 0, stream>>>(wgt, wt);
        const long totalX = (long)T_STEPS * BATCH * I_DIM;
        prep_x<<<(int)((totalX + 255) / 256), 256, 0, stream>>>(x, xs);
        prep_bias<<<(N_BR * F4 + 255) / 256, 256, 0, stream>>>(bi, bh, bias);
        const int stCount = N_BR * BATCH * H_DIM;
        prep_zero<<<(stCount + 255) / 256, 256, 0, stream>>>(hs, cst, stCount);
    }

    // Sequential recurrence: one launch per timestep (launch order = sync)
    for (int t = 0; t < T_STEPS; ++t) {
        lstm_step<<<64, 256, 0, stream>>>(xs, dur, wt, bias, hs, cst, out, t);
    }
}
